// RefactAttention_77446850281606
// MI455X (gfx1250) — matrix operand order repaired and checked by reading, not compiled, not run on hardware
//
#include <hip/hip_runtime.h>
#include <hip/hip_bf16.h>

// ---------------------------------------------------------------------------
// MQA attention block for MI455X (gfx1250): bf16 WMMA + Tensor Data Mover.
//   q  = (hs @ Wq) * 1/sqrt(128)          -> bf16
//   kv = hs @ Wkv ; k = kv[:,:128], v = kv[:,128:]
//   attn = softmax(causal(q k^T + alibi)) -> flash, online softmax
//   out  = (attn @ v) @ Wo                -> f32
// ---------------------------------------------------------------------------

#define HIDDEN   4096
#define NHEADS   32
#define HDIM     128
#define BATCH    2
#define SEQ      1024

typedef __bf16 bf16;
typedef __attribute__((ext_vector_type(4)))  float v4f;
typedef __attribute__((ext_vector_type(8)))  float v8f;
typedef __attribute__((ext_vector_type(4)))  bf16  v4bf;
typedef __attribute__((ext_vector_type(8)))  bf16  v8bf;
typedef __attribute__((ext_vector_type(16))) bf16  v16bf;
typedef __attribute__((ext_vector_type(4)))  unsigned int u32x4;
typedef __attribute__((ext_vector_type(4)))  int i32x4;
typedef __attribute__((ext_vector_type(8)))  int i32x8;

#if __has_builtin(__builtin_amdgcn_tensor_load_to_lds) && \
    __has_builtin(__builtin_amdgcn_s_wait_tensorcnt)
#define HAVE_TDM 1
#else
#define HAVE_TDM 0
#endif

__device__ __forceinline__ v16bf cat16(v8bf lo, v8bf hi) {
  return __builtin_shufflevector(lo, hi, 0,1,2,3,4,5,6,7,8,9,10,11,12,13,14,15);
}

// xor-butterfly reductions across the 16-lane half-groups of a wave32
// (matches the WMMA C-layout: row = v + 8*half, col = lane&15).
__device__ __forceinline__ float half16_max(float x) {
  x = fmaxf(x, __int_as_float(__builtin_amdgcn_ds_swizzle(__float_as_int(x), 0x041f)));
  x = fmaxf(x, __int_as_float(__builtin_amdgcn_ds_swizzle(__float_as_int(x), 0x081f)));
  x = fmaxf(x, __int_as_float(__builtin_amdgcn_ds_swizzle(__float_as_int(x), 0x101f)));
  x = fmaxf(x, __int_as_float(__builtin_amdgcn_ds_swizzle(__float_as_int(x), 0x201f)));
  return x;
}
__device__ __forceinline__ float half16_sum(float x) {
  x += __int_as_float(__builtin_amdgcn_ds_swizzle(__float_as_int(x), 0x041f));
  x += __int_as_float(__builtin_amdgcn_ds_swizzle(__float_as_int(x), 0x081f));
  x += __int_as_float(__builtin_amdgcn_ds_swizzle(__float_as_int(x), 0x101f));
  x += __int_as_float(__builtin_amdgcn_ds_swizzle(__float_as_int(x), 0x201f));
  return x;
}

#if HAVE_TDM
// ---------------------------------------------------------------------------
// Tensor Data Mover: 2D tile load Global->LDS with per-row LDS padding.
//   tile_d0  : elements per row (bf16, data_size code 1)
//   rows     : number of rows
//   stride   : tensor dim0 stride in elements
//   pad_iv   : pad_interval code (row bytes = 2^(code+3))
//   pad_amt  : pad_amount code (pad dwords = code+1)
// Issued by one wave; caller waits TENSORcnt then barriers.
// ---------------------------------------------------------------------------
__device__ __forceinline__ void tdm_load_2d(unsigned lds_off, const void* gptr,
                                            unsigned tile_d0, unsigned rows,
                                            unsigned stride,
                                            unsigned pad_iv, unsigned pad_amt) {
  const unsigned long long ga = (unsigned long long)gptr;
  u32x4 g0;
  g0[0] = 1u;                                      // count=1, user mode
  g0[1] = lds_off;                                 // LDS byte address
  g0[2] = (unsigned)(ga & 0xffffffffu);            // global addr [31:0]
  g0[3] = (unsigned)((ga >> 32) & 0x01ffffffu)     // global addr [56:32]
        | (2u << 30);                              // type = 2 ("image")
  i32x8 g1;
  g1[0] = (int)((1u << 16)                         // data_size = 2 bytes
              | (1u << 20)                         // pad_enable
              | (pad_iv << 22) | (pad_amt << 25));
  g1[1] = (int)(tile_d0 << 16);                    // tensor_dim0[15:0]
  g1[2] = (int)((tile_d0 >> 16) | (rows << 16));   // dim0[31:16] | dim1[15:0]
  g1[3] = (int)((rows >> 16) | (tile_d0 << 16));   // dim1[31:16] | tile_dim0
  g1[4] = (int)(rows & 0xffffu);                   // tile_dim1 | tile_dim2=0
  g1[5] = (int)stride;                             // dim0_stride[31:0]
  g1[6] = 0;                                       // stride hi | dim1_stride lo
  g1[7] = 0;
  const i32x4 z = {0, 0, 0, 0};
#if defined(__clang_major__) && __clang_major__ >= 23
  const i32x8 z8 = {};
  __builtin_amdgcn_tensor_load_to_lds(g0, g1, z, z, z8, 0);
#else
  __builtin_amdgcn_tensor_load_to_lds(g0, g1, z, z, 0);
#endif
}
__device__ __forceinline__ unsigned lds_offset(const void* p) {
  return (unsigned)(unsigned long long)p;   // flat LDS addr: low 32 bits = LDS offset
}
#endif

// ---------------------------------------------------------------------------
// f32 -> bf16 downconvert (vectorized x4, grid-stride)
// ---------------------------------------------------------------------------
__global__ void cvt_f32_to_bf16(const float* __restrict__ s, bf16* __restrict__ d, long n4) {
  long i  = (long)blockIdx.x * blockDim.x + threadIdx.x;
  long st = (long)gridDim.x * blockDim.x;
  for (; i < n4; i += st) {
    v4f x = ((const v4f*)s)[i];
    v4bf y;
    y[0] = (bf16)x[0]; y[1] = (bf16)x[1]; y[2] = (bf16)x[2]; y[3] = (bf16)x[3];
    ((v4bf*)d)[i] = y;
  }
}

// ---------------------------------------------------------------------------
// bf16 GEMM: C[M,N] = alpha * A[M,K] @ B[K,N].  A,B bf16 row-major.
// Block = 256 threads (8 waves), tile 128x128, BK=32.
// Wave grid 4x2: each wave owns 32 rows x 64 cols = 8 WMMAs per K-step.
// A tile staged by the TDM (padding gives the stride-40 LDS layout directly);
// B tile staged transposed so fragment reads are contiguous ds_load_b128s.
// ---------------------------------------------------------------------------
template <bool OUT_BF16>
__global__ __launch_bounds__(256) void gemm_bf16_wmma(
    const bf16* __restrict__ A, const bf16* __restrict__ B, void* __restrict__ C,
    int M, int N, int K, float alpha)
{
  __shared__ __attribute__((aligned(16))) bf16 As[128 * 40];   // 128r x 32k, stride 40
  __shared__ __attribute__((aligned(16))) bf16 BsT[128 * 40];  // 128n x 32k, transposed

  const int t    = threadIdx.x;
  const int lane = t & 31, w = t >> 5;
  const int half = lane >> 4, l16 = lane & 15;
  const int row0 = blockIdx.y * 128;
  const int col0 = blockIdx.x * 128;
  const int rw = (w >> 1) * 32;        // wave row offset within tile
  const int cw = (w & 1) * 64;         // wave col offset within tile

  v8f acc[2][4] = {};

  const int bk = t >> 3, bc = (t & 7) * 16;   // B staging: 16 bf16 per thread
#if !HAVE_TDM
  const int ar = t >> 1, ac = (t & 1) * 16;   // A staging fallback
#endif

  for (int k0 = 0; k0 < K; k0 += 32) {
    __syncthreads();
#if HAVE_TDM
    if (w == 0) {  // one wave drives the DMA of the whole 128x32 A tile
      tdm_load_2d(lds_offset(As), A + (size_t)row0 * K + k0,
                  /*tile_d0=*/32, /*rows=*/128, /*stride=*/(unsigned)K,
                  /*pad_iv: 64B rows=*/3, /*pad: 4 dwords=*/3);
      __builtin_amdgcn_s_wait_tensorcnt(0);
    }
#else
    {
      const v8bf* asrc = (const v8bf*)(A + (size_t)(row0 + ar) * K + k0 + ac);
      *(v8bf*)&As[ar * 40 + ac]     = asrc[0];
      *(v8bf*)&As[ar * 40 + ac + 8] = asrc[1];
    }
#endif
    { // stage B tile transposed: BsT[n][k]
      const bf16* bsrc = B + (size_t)(k0 + bk) * N + col0 + bc;
      v8bf b0 = *(const v8bf*)(bsrc);
      v8bf b1 = *(const v8bf*)(bsrc + 8);
#pragma unroll
      for (int e = 0; e < 8; ++e) {
        BsT[(bc + e) * 40 + bk]     = b0[e];
        BsT[(bc + 8 + e) * 40 + bk] = b1[e];
      }
      if (k0 + 32 < K)  // gfx1250 global_prefetch_b8 for next B tile
        __builtin_prefetch(bsrc + (size_t)32 * N, 0, 3);
    }
    __syncthreads();

#pragma unroll
    for (int sub = 0; sub < 2; ++sub) {
      // A fragment 16x32: lane row = l16; elems 0..7 -> K=half*8+j, 8..15 -> 16+half*8+j
      const int r = rw + sub * 16 + l16;
      v16bf afrag = cat16(*(const v8bf*)&As[r * 40 + half * 8],
                          *(const v8bf*)&As[r * 40 + half * 8 + 16]);
#pragma unroll
      for (int nt = 0; nt < 4; ++nt) {
        const int c = cw + nt * 16 + l16;   // B fragment: elems 0..7 -> K = half*8 + j, 8..15 -> 16 + half*8 + j
        v16bf bfrag = cat16(*(const v8bf*)&BsT[c * 40 + half * 8],
                            *(const v8bf*)&BsT[c * 40 + half * 8 + 16]);
        acc[sub][nt] = __builtin_amdgcn_wmma_f32_16x16x32_bf16(
            false, afrag, false, bfrag, (short)0, acc[sub][nt], false, false);
      }
    }
  }

  // epilogue: C layout row = v + 8*half, col = lane&15
#pragma unroll
  for (int sub = 0; sub < 2; ++sub)
#pragma unroll
    for (int nt = 0; nt < 4; ++nt)
#pragma unroll
      for (int v = 0; v < 8; ++v) {
        const size_t row = (size_t)(row0 + rw + sub * 16 + v + 8 * half);
        const size_t col = (size_t)(col0 + cw + nt * 16 + l16);
        const float val = acc[sub][nt][v] * alpha;
        if (OUT_BF16) ((bf16*)C)[row * (size_t)N + col] = (bf16)val;
        else          ((float*)C)[row * (size_t)N + col] = val;
      }
}

// ---------------------------------------------------------------------------
// Flash attention with ALiBi + causal mask, bf16 WMMA, online softmax.
// grid = (SEQ/128, NHEADS, BATCH); block = 256 (8 waves x 16 query rows).
// The 32x256 KV chunk is staged in LDS by ONE TDM transfer (K fragments read
// straight out of the padded TDM destination; V is transposed LDS->LDS).
// P round-trips through per-wave LDS to convert C-layout -> A-layout.
// Q is pre-scaled by 1/sqrt(HDIM) at projection time.
// ---------------------------------------------------------------------------
__global__ __launch_bounds__(256) void attn_wmma(
    const bf16* __restrict__ Qb,   // [B*S, HIDDEN]: (b*S+q)*4096 + h*128 + d
    const bf16* __restrict__ KVb,  // [B*S, 256]: K = [:,0:128], V = [:,128:256]
    bf16* __restrict__ AOb)        // [B*S, HIDDEN] attention output, bf16
{
  __shared__ __attribute__((aligned(16))) bf16 KVs[32 * 264];   // 32 keys x 256, stride 264
  __shared__ __attribute__((aligned(16))) bf16 VsT[128 * 40];   // 128 d x 32 keys
  __shared__ __attribute__((aligned(16))) bf16 Ps[8 * 16 * 40]; // per-wave P: 16q x 32k

  const int t    = threadIdx.x;
  const int lane = t & 31, w = t >> 5;
  const int half = lane >> 4, l16 = lane & 15;
  const int qbase = blockIdx.x * 128;
  const int h  = blockIdx.y;
  const int bz = blockIdx.z;
  const int q0 = qbase + w * 16;
  // alibi slope: start = 2^(-2^-(log2(32)-3)) = 2^-0.25 ; slope_h = start^(h+1)
  const float slope = exp2f(-0.25f * (float)(h + 1));

  // Q fragments: 16 rows x 128 d = 4 A-fragments (Q already holds 1/sqrt(d))
  v16bf qf[4];
  {
    const int qrow = q0 + l16;
    const bf16* qp = Qb + (size_t)(bz * SEQ + qrow) * HIDDEN + h * HDIM;
#pragma unroll
    for (int dc = 0; dc < 4; ++dc)
      qf[dc] = cat16(*(const v8bf*)(qp + dc * 32 + half * 8),
                     *(const v8bf*)(qp + dc * 32 + half * 8 + 16));
  }

  v8f acc[8] = {};
  float m[8], l[8];
#pragma unroll
  for (int v = 0; v < 8; ++v) { m[v] = -__builtin_inff(); l[v] = 0.f; }

  const int kr = t >> 3, kc = (t & 7) * 16;  // staging indices

  for (int kb = 0; kb < qbase + 128; kb += 32) {
    __syncthreads();   // KVs/VsT free to overwrite
#if HAVE_TDM
    if (w == 0) {      // DMA the whole 32x256 KV chunk; pad -> stride 264
      tdm_load_2d(lds_offset(KVs), KVb + (size_t)(bz * SEQ + kb) * (2 * HDIM),
                  /*tile_d0=*/256, /*rows=*/32, /*stride=*/256,
                  /*pad_iv: 512B rows=*/6, /*pad: 4 dwords=*/3);
      __builtin_amdgcn_s_wait_tensorcnt(0);
    }
#else
    { // manual staging: 32 bf16 per thread
      const bf16* kvrow = KVb + (size_t)(bz * SEQ + kb + kr) * (2 * HDIM);
      const int c2 = (t & 7) * 32;
      *(v8bf*)&KVs[kr * 264 + c2]      = *(const v8bf*)(kvrow + c2);
      *(v8bf*)&KVs[kr * 264 + c2 + 8]  = *(const v8bf*)(kvrow + c2 + 8);
      *(v8bf*)&KVs[kr * 264 + c2 + 16] = *(const v8bf*)(kvrow + c2 + 16);
      *(v8bf*)&KVs[kr * 264 + c2 + 24] = *(const v8bf*)(kvrow + c2 + 24);
    }
#endif
    __syncthreads();   // KVs ready
    { // transpose V half of the chunk: VsT[d][k]
      v8bf v0 = *(const v8bf*)&KVs[kr * 264 + HDIM + kc];
      v8bf v1 = *(const v8bf*)&KVs[kr * 264 + HDIM + kc + 8];
#pragma unroll
      for (int e = 0; e < 8; ++e) {
        VsT[(kc + e) * 40 + kr]     = v0[e];
        VsT[(kc + 8 + e) * 40 + kr] = v1[e];
      }
    }
    __syncthreads();   // VsT ready
    if (kb > q0 + 15) continue;   // fully above causal frontier for this wave

    // ---- S = Q K^T (two 16x16 score tiles, 4 WMMAs each over d) ----
    v8f s[2];
#pragma unroll
    for (int kt = 0; kt < 2; ++kt) {
      v8f sc = {};
      const int kk = kt * 16 + l16;
#pragma unroll
      for (int dc = 0; dc < 4; ++dc) {
        v16bf kf = cat16(*(const v8bf*)&KVs[kk * 264 + dc * 32 + half * 8],
                         *(const v8bf*)&KVs[kk * 264 + dc * 32 + half * 8 + 16]);
        sc = __builtin_amdgcn_wmma_f32_16x16x32_bf16(
            false, qf[dc], false, kf, (short)0, sc, false, false);
      }
      const int kpos = kb + kk;
#pragma unroll
      for (int v = 0; v < 8; ++v) {
        const int qpos = q0 + v + 8 * half;
        sc[v] = (kpos <= qpos) ? sc[v] + slope * (float)(kpos - qpos)
                               : -__builtin_inff();
      }
      s[kt] = sc;
    }

    // ---- online softmax update + stash P (bf16) into per-wave LDS ----
#pragma unroll
    for (int v = 0; v < 8; ++v) {
      const float rm = half16_max(fmaxf(s[0][v], s[1][v]));
      const float mn = fmaxf(m[v], rm);
      const float co = expf(m[v] - mn);           // 0 on first chunk (m = -inf)
      const float p0 = expf(s[0][v] - mn);
      const float p1 = expf(s[1][v] - mn);
      const float rs = half16_sum(p0 + p1);
      l[v] = l[v] * co + rs;
      m[v] = mn;
#pragma unroll
      for (int dt = 0; dt < 8; ++dt) acc[dt][v] *= co;
      const int r = v + 8 * half;                 // C-layout row
      Ps[(w * 16 + r) * 40 + l16]      = (bf16)p0;
      Ps[(w * 16 + r) * 40 + 16 + l16] = (bf16)p1;
    }

    // ---- O += P V : re-read P in A-layout, V fragments from transposed LDS ----
    v16bf pf = cat16(*(const v8bf*)&Ps[(w * 16 + l16) * 40 + half * 8],
                     *(const v8bf*)&Ps[(w * 16 + l16) * 40 + half * 8 + 16]);
#pragma unroll
    for (int dt = 0; dt < 8; ++dt) {
      const int dcol = dt * 16 + l16;
      v16bf vf = cat16(*(const v8bf*)&VsT[dcol * 40 + half * 8],
                       *(const v8bf*)&VsT[dcol * 40 + half * 8 + 16]);
      acc[dt] = __builtin_amdgcn_wmma_f32_16x16x32_bf16(
          false, pf, false, vf, (short)0, acc[dt], false, false);
    }
  }

  // ---- finalize: divide by l, write bf16 in (b, q, h*128+d) layout ----
#pragma unroll
  for (int dt = 0; dt < 8; ++dt)
#pragma unroll
    for (int v = 0; v < 8; ++v) {
      const int qpos = q0 + v + 8 * half;
      const float o = acc[dt][v] / l[v];
      AOb[(size_t)(bz * SEQ + qpos) * HIDDEN + h * HDIM + dt * 16 + l16] = (bf16)o;
    }
}

// ---------------------------------------------------------------------------
// Host orchestration.
// Workspace layout (bytes): hsb 16.8M | wqb 33.6M | wkvb 2.1M | wob 33.6M |
//                           qb 16.8M | kvb 1.0M | aob 16.8M  (~121 MB total)
// ---------------------------------------------------------------------------
extern "C" void kernel_launch(void* const* d_in, const int* in_sizes, int n_in,
                              void* d_out, int out_size, void* d_ws, size_t ws_size,
                              hipStream_t stream)
{
  (void)in_sizes; (void)n_in; (void)out_size; (void)ws_size;
  const float* hs  = (const float*)d_in[0];
  const float* Wq  = (const float*)d_in[1];
  const float* Wkv = (const float*)d_in[2];
  const float* Wo  = (const float*)d_in[3];

  const size_t M = (size_t)BATCH * SEQ;  // 2048

  char* p = (char*)d_ws;
  bf16* hsb  = (bf16*)p; p += M * HIDDEN * 2;
  bf16* wqb  = (bf16*)p; p += (size_t)HIDDEN * HIDDEN * 2;
  bf16* wkvb = (bf16*)p; p += (size_t)HIDDEN * (2 * HDIM) * 2;
  bf16* wob  = (bf16*)p; p += (size_t)HIDDEN * HIDDEN * 2;
  bf16* qb   = (bf16*)p; p += M * HIDDEN * 2;
  bf16* kvb  = (bf16*)p; p += M * (2 * HDIM) * 2;
  bf16* aob  = (bf16*)p; p += M * HIDDEN * 2;

  auto cvt = [&](const float* s, bf16* d, size_t n) {
    long n4 = (long)(n / 4);
    int blocks = (int)((n4 + 255) / 256);
    if (blocks > 4096) blocks = 4096;
    cvt_f32_to_bf16<<<dim3(blocks), dim3(256), 0, stream>>>(s, d, n4);
  };
  cvt(hs,  hsb,  M * HIDDEN);
  cvt(Wq,  wqb,  (size_t)HIDDEN * HIDDEN);
  cvt(Wkv, wkvb, (size_t)HIDDEN * (2 * HDIM));
  cvt(Wo,  wob,  (size_t)HIDDEN * HIDDEN);

  const float scale = 0.08838834764831845f;  // 1/sqrt(128), folded into Q

  // Q = (hs @ Wq) * scale -> bf16
  gemm_bf16_wmma<true><<<dim3(HIDDEN / 128, (int)(M / 128)), dim3(256), 0, stream>>>(
      hsb, wqb, (void*)qb, (int)M, HIDDEN, HIDDEN, scale);

  // KV = hs @ Wkv -> bf16
  gemm_bf16_wmma<true><<<dim3((2 * HDIM) / 128, (int)(M / 128)), dim3(256), 0, stream>>>(
      hsb, wkvb, (void*)kvb, (int)M, 2 * HDIM, HIDDEN, 1.0f);

  // flash attention
  attn_wmma<<<dim3(SEQ / 128, NHEADS, BATCH), dim3(256), 0, stream>>>(qb, kvb, aob);

  // out = attn_out @ Wo -> f32 into d_out
  gemm_bf16_wmma<false><<<dim3(HIDDEN / 128, (int)(M / 128)), dim3(256), 0, stream>>>(
      aob, wob, d_out, (int)M, HIDDEN, HIDDEN, 1.0f);
}
